// SparseMLP_45002667327744
// MI455X (gfx1250) — matrix operand order repaired and checked by reading, not compiled, not run on hardware
//
#include <hip/hip_runtime.h>

// ---------------------------------------------------------------------------
// MoE MLP for MI455X (gfx1250, wave32):
//   h   = gelu_tanh(x_e @ w1_e^T)   (f32 in, bf16 out to workspace)
//   out = h @ w2_e                  (bf16 A via TDM, f32->bf16 B, f32 out)
// AI ~820 flop/byte -> compute bound -> V_WMMA_F32_16X16X32_BF16.
// Block tile 128x128, 8 waves (4Mx2N), wave tile 32x64 = 8 WMMA / K-step.
// GEMM2 stages its bf16 A tiles with the Tensor Data Mover (TENSORcnt),
// double-buffered LDS, one barrier per K-step.
// ---------------------------------------------------------------------------

typedef __attribute__((ext_vector_type(4)))  __bf16 v4bf;
typedef __attribute__((ext_vector_type(8)))  __bf16 v8bf;
typedef __attribute__((ext_vector_type(16))) __bf16 v16bf;
typedef __attribute__((ext_vector_type(8)))  float  v8f;
typedef __attribute__((ext_vector_type(4)))  unsigned int v4u;
typedef __attribute__((ext_vector_type(4)))  int  v4i;
typedef __attribute__((ext_vector_type(8)))  int  v8i;

#define BM 128          // block tile M
#define BN 128          // block tile N
#define BK 32           // K step == one WMMA K
#define LDS_PITCH 40    // BK + 8 bf16 pad (80B row stride, 16B aligned)

#if __has_builtin(__builtin_amdgcn_tensor_load_to_lds)
#define HAVE_TDM 1
#else
#define HAVE_TDM 0
#endif

__device__ __forceinline__ float gelu_tanh(float x) {
    const float k0 = 0.7978845608028654f;   // sqrt(2/pi)
    const float k1 = 0.044715f;
    return 0.5f * x * (1.0f + tanhf(k0 * (x + k1 * x * x * x)));
}

__device__ __forceinline__ v16bf cat8(v8bf lo, v8bf hi) {
    return __builtin_shufflevector(lo, hi,
        0, 1, 2, 3, 4, 5, 6, 7, 8, 9, 10, 11, 12, 13, 14, 15);
}

__device__ __forceinline__ v4bf cvt4(float4 v) {
    v4bf p = { (__bf16)v.x, (__bf16)v.y, (__bf16)v.z, (__bf16)v.w };
    return p;
}

__device__ __forceinline__ void wait_tensorcnt0() {
#if __has_builtin(__builtin_amdgcn_s_wait_tensorcnt)
    __builtin_amdgcn_s_wait_tensorcnt(0);
#else
    asm volatile("s_wait_tensorcnt 0x0" ::: "memory");
#endif
}

#if HAVE_TDM
// TDM: DMA a [128 rows x 32 bf16] tile (row stride Fdim elements) from global
// into LDS with an 80B padded pitch (pad 16B after each 64B row).
// D# layout per CDNA5 ISA.
// This toolchain exposes the 6-arg clang-23 builtin:
//   (uint32x4 g0, int32x8 g1, int32x4 g2, int32x4 g3, int32x8, i32 cpol)
__device__ __forceinline__ void tdm_load_a_tile(unsigned lds_addr,
                                                const __bf16* gsrc,
                                                int Fdim, int Trows) {
    unsigned long long ga = (unsigned long long)(uintptr_t)gsrc;
    v4u g0;
    g0[0] = 1u;                                           // count=1, user mode
    g0[1] = lds_addr;                                     // LDS byte address
    g0[2] = (unsigned)(ga & 0xffffffffull);               // global_addr[31:0]
    g0[3] = (unsigned)((ga >> 32) & 0x01ffffffull)        // global_addr[56:32]
          | (2u << 30);                                   // type = 2 ("image")
    v8i g1;
    g1[0] = (int)((1u << 16)      // data_size = 1 -> 2 bytes
                | (1u << 20)      // pad_enable
                | (3u << 22)      // pad_interval code 3 -> every 16 DW (64B)
                | (3u << 25));    // pad_amount  code 3 -> 4 DW (16B)
    unsigned d0 = (unsigned)Fdim;         // tensor_dim0 (elements)
    unsigned d1 = (unsigned)Trows;        // tensor_dim1 (rows)
    g1[1] = (int)((d0 & 0xffffu) << 16);                  // [63:48] dim0 lo16
    g1[2] = (int)((d0 >> 16) | ((d1 & 0xffffu) << 16));   // dim0 hi16 | dim1 lo16
    g1[3] = (int)((d1 >> 16) | ((unsigned)BK << 16));     // dim1 hi16 | tile_dim0
    g1[4] = (int)(unsigned)BM;                            // tile_dim1 | tile_dim2=0
    g1[5] = (int)(unsigned)Fdim;                          // dim0_stride[31:0]
    g1[6] = 0;                                            // stride hi | dim1_stride lo
    g1[7] = 0;
    v4i z4 = {0, 0, 0, 0};                                // 2D: groups 2/3 unused
    v8i z8 = {0, 0, 0, 0, 0, 0, 0, 0};
    __builtin_amdgcn_tensor_load_to_lds(g0, g1, z4, z4, z8, 0);
}
#endif

// ---------------------------------------------------------------------------
// GEMM1: C[T, F] = X[T, H] @ W1[e][F, H]^T, gelu, bf16 store.
// ---------------------------------------------------------------------------
__global__ __launch_bounds__(256) void moe_gemm1_gelu(
    const float* __restrict__ X,      // [T, H]
    const float* __restrict__ W1,     // [E, F, H]
    __bf16* __restrict__ Hws,         // [T, F] bf16
    int Hdim, int Fdim, int cap)
{
    __shared__ __bf16 As[BM][LDS_PITCH];
    __shared__ __bf16 Bs[BN][LDS_PITCH];

    const int tid  = threadIdx.x;
    const int lane = tid & 31;
    const int wave = tid >> 5;
    const int wm   = wave >> 1;
    const int wn   = wave & 1;

    const int row0 = blockIdx.y * BM;
    const int col0 = blockIdx.x * BN;
    const int e    = row0 / cap;
    const float* Wsrc = W1 + (size_t)e * Fdim * Hdim;

    const int r    = lane & 15;
    const int hsel = lane >> 4;
    const int kbA  = hsel * 8;
    const int kbB  = hsel * 16;

    const int sRow = tid >> 3;
    const int sCol = (tid & 7) << 2;

    const float* aPtr = X    + (size_t)(row0 + sRow) * Hdim + sCol;
    const float* bPtr = Wsrc + (size_t)(col0 + sRow) * Hdim + sCol;
    const size_t ldG  = (size_t)32 * Hdim;

    float4 aReg[4], bReg[4];
    #pragma unroll
    for (int it = 0; it < 4; ++it) {
        aReg[it] = *reinterpret_cast<const float4*>(aPtr + it * ldG);
        bReg[it] = *reinterpret_cast<const float4*>(bPtr + it * ldG);
        __builtin_prefetch(aPtr + it * ldG + BK, 0, 0);
        __builtin_prefetch(bPtr + it * ldG + BK, 0, 0);
    }

    v8f acc[2][4] = {};

    for (int k0 = 0; k0 < Hdim; k0 += BK) {
        __syncthreads();
        #pragma unroll
        for (int it = 0; it < 4; ++it) {
            *reinterpret_cast<v4bf*>(&As[sRow + it * 32][sCol]) = cvt4(aReg[it]);
            *reinterpret_cast<v4bf*>(&Bs[sRow + it * 32][sCol]) = cvt4(bReg[it]);
        }
        if (k0 + BK < Hdim) {
            const float* aN = aPtr + (k0 + BK);
            const float* bN = bPtr + (k0 + BK);
            #pragma unroll
            for (int it = 0; it < 4; ++it) {
                aReg[it] = *reinterpret_cast<const float4*>(aN + it * ldG);
                bReg[it] = *reinterpret_cast<const float4*>(bN + it * ldG);
                __builtin_prefetch(aN + it * ldG + BK, 0, 0);
                __builtin_prefetch(bN + it * ldG + BK, 0, 0);
            }
        }
        __syncthreads();

        v16bf a[2], b[4];
        #pragma unroll
        for (int i = 0; i < 2; ++i) {
            const int arow = wm * 32 + i * 16 + r;
            a[i] = cat8(*reinterpret_cast<const v8bf*>(&As[arow][kbA]),
                        *reinterpret_cast<const v8bf*>(&As[arow][kbA + 16]));
        }
        #pragma unroll
        for (int j = 0; j < 4; ++j) {
            const int brow = wn * 64 + j * 16 + r;
            b[j] = cat8(*reinterpret_cast<const v8bf*>(&Bs[brow][kbA]),
                        *reinterpret_cast<const v8bf*>(&Bs[brow][kbA + 16]));
        }
        #pragma unroll
        for (int i = 0; i < 2; ++i)
            #pragma unroll
            for (int j = 0; j < 4; ++j)
                acc[i][j] = __builtin_amdgcn_wmma_f32_16x16x32_bf16(
                    false, a[i], false, b[j], (short)0, acc[i][j], false, false);
    }

    #pragma unroll
    for (int i = 0; i < 2; ++i) {
        #pragma unroll
        for (int j = 0; j < 4; ++j) {
            const int rg0 = row0 + wm * 32 + i * 16 + hsel * 8;
            const int cg  = col0 + wn * 64 + j * 16 + r;
            #pragma unroll
            for (int v = 0; v < 8; ++v)
                Hws[(size_t)(rg0 + v) * Fdim + cg] = (__bf16)gelu_tanh(acc[i][j][v]);
        }
    }
}

// ---------------------------------------------------------------------------
// GEMM2: Out[T, H] = Hws[T, F](bf16) @ W2[e][F, H].
// A tiles DMA'd by the Tensor Data Mover; B transposed+converted via VGPRs.
// Double-buffered LDS, one barrier per K-step.
// ---------------------------------------------------------------------------
__global__ __launch_bounds__(256) void moe_gemm2(
    const __bf16* __restrict__ Hws,   // [T, F] bf16
    const float*  __restrict__ W2,    // [E, F, H]
    float* __restrict__ Out,          // [T, H]
    int Fdim, int Hdim, int cap, int Trows)
{
    __shared__ __bf16 As[2][BM][LDS_PITCH];
    __shared__ __bf16 Bs[2][BN][LDS_PITCH];

    const int tid  = threadIdx.x;
    const int lane = tid & 31;
    const int wave = tid >> 5;
    const int wm   = wave >> 1;
    const int wn   = wave & 1;

    const int row0 = blockIdx.y * BM;
    const int col0 = blockIdx.x * BN;
    const int e    = row0 / cap;
    const float* Wsrc = W2 + (size_t)e * Fdim * Hdim;

    const int r    = lane & 15;
    const int hsel = lane >> 4;
    const int kbA  = hsel * 8;
    const int kbB  = hsel * 16;

    // B staging: Bs[n][k] = W2[k0+k][col0+n]; 4 float4 per thread along N
    const int bK  = tid >> 5;               // 0..7 (k advances by 8 per it)
    const int bN4 = (tid & 31) << 2;        // 0..124
    const float* bPtr = Wsrc + (size_t)bK * Hdim + col0 + bN4;
    const size_t bLdG = (size_t)8 * Hdim;

    const __bf16* aBase = Hws + (size_t)row0 * Fdim;

#if !HAVE_TDM
    const int aRow = tid >> 2;
    const int aCol = (tid & 3) << 3;
    const __bf16* aPtr = aBase + (size_t)aRow * Fdim + aCol;
    const size_t aLdG  = (size_t)64 * Fdim;
    v8bf aReg[2];
#endif

    float4 bReg[4];

    // ---- prologue: fill buffer 0 for k0 = 0 ----
#if HAVE_TDM
    if (wave == 0)
        tdm_load_a_tile((unsigned)(uintptr_t)&As[0][0][0], aBase, Fdim, Trows);
#else
    #pragma unroll
    for (int it = 0; it < 2; ++it)
        aReg[it] = *reinterpret_cast<const v8bf*>(aPtr + it * aLdG);
#endif
    #pragma unroll
    for (int it = 0; it < 4; ++it) {
        bReg[it] = *reinterpret_cast<const float4*>(bPtr + it * bLdG);
        __builtin_prefetch(bPtr + it * bLdG + (size_t)BK * Hdim, 0, 0);
    }
    #pragma unroll
    for (int it = 0; it < 4; ++it) {
        float4 v = bReg[it];
        const int k = bK + it * 8;
        Bs[0][bN4 + 0][k] = (__bf16)v.x;
        Bs[0][bN4 + 1][k] = (__bf16)v.y;
        Bs[0][bN4 + 2][k] = (__bf16)v.z;
        Bs[0][bN4 + 3][k] = (__bf16)v.w;
    }
#if !HAVE_TDM
    #pragma unroll
    for (int it = 0; it < 2; ++it)
        *reinterpret_cast<v8bf*>(&As[0][aRow + it * 64][aCol]) = aReg[it];
#endif
#if HAVE_TDM
    if (wave == 0) wait_tensorcnt0();
#endif
    __syncthreads();

    v8f acc[2][4] = {};
    int p = 0;

    for (int k0 = 0; k0 < Fdim; k0 += BK, p ^= 1) {
        const bool more = (k0 + BK < Fdim);
        if (more) {
            // kick off next tile into the idle buffer while we compute
#if HAVE_TDM
            if (wave == 0)
                tdm_load_a_tile((unsigned)(uintptr_t)&As[p ^ 1][0][0],
                                aBase + (k0 + BK), Fdim, Trows);
#else
            const __bf16* aN = aPtr + (k0 + BK);
            #pragma unroll
            for (int it = 0; it < 2; ++it)
                aReg[it] = *reinterpret_cast<const v8bf*>(aN + it * aLdG);
#endif
            const float* bNp = bPtr + (size_t)(k0 + BK) * Hdim;
            #pragma unroll
            for (int it = 0; it < 4; ++it) {
                bReg[it] = *reinterpret_cast<const float4*>(bNp + it * bLdG);
                __builtin_prefetch(bNp + it * bLdG + (size_t)BK * Hdim, 0, 0);
            }
        }

        // ---- compute on buffer p ----
        v16bf a[2], b[4];
        #pragma unroll
        for (int i = 0; i < 2; ++i) {
            const int arow = wm * 32 + i * 16 + r;
            a[i] = cat8(*reinterpret_cast<const v8bf*>(&As[p][arow][kbA]),
                        *reinterpret_cast<const v8bf*>(&As[p][arow][kbA + 16]));
        }
        #pragma unroll
        for (int j = 0; j < 4; ++j) {
            const int brow = wn * 64 + j * 16 + r;
            b[j] = cat8(*reinterpret_cast<const v8bf*>(&Bs[p][brow][kbA]),
                        *reinterpret_cast<const v8bf*>(&Bs[p][brow][kbA + 16]));
        }
        #pragma unroll
        for (int i = 0; i < 2; ++i)
            #pragma unroll
            for (int j = 0; j < 4; ++j)
                acc[i][j] = __builtin_amdgcn_wmma_f32_16x16x32_bf16(
                    false, a[i], false, b[j], (short)0, acc[i][j], false, false);

        if (more) {
            // write staged B into the idle buffer, then publish both tiles
            #pragma unroll
            for (int it = 0; it < 4; ++it) {
                float4 v = bReg[it];
                const int k = bK + it * 8;
                Bs[p ^ 1][bN4 + 0][k] = (__bf16)v.x;
                Bs[p ^ 1][bN4 + 1][k] = (__bf16)v.y;
                Bs[p ^ 1][bN4 + 2][k] = (__bf16)v.z;
                Bs[p ^ 1][bN4 + 3][k] = (__bf16)v.w;
            }
#if !HAVE_TDM
            #pragma unroll
            for (int it = 0; it < 2; ++it)
                *reinterpret_cast<v8bf*>(&As[p ^ 1][aRow + it * 64][aCol]) = aReg[it];
#endif
#if HAVE_TDM
            if (wave == 0) wait_tensorcnt0();
#endif
            __syncthreads();
        }
    }

    #pragma unroll
    for (int i = 0; i < 2; ++i) {
        #pragma unroll
        for (int j = 0; j < 4; ++j) {
            const int rg0 = row0 + wm * 32 + i * 16 + hsel * 8;
            const int cg  = col0 + wn * 64 + j * 16 + r;
            #pragma unroll
            for (int v = 0; v < 8; ++v)
                Out[(size_t)(rg0 + v) * Hdim + cg] = acc[i][j][v];
        }
    }
}

extern "C" void kernel_launch(void* const* d_in, const int* in_sizes, int n_in,
                              void* d_out, int out_size, void* d_ws, size_t ws_size,
                              hipStream_t stream) {
    (void)in_sizes; (void)n_in; (void)out_size; (void)ws_size;
    const float* x  = (const float*)d_in[0];   // [T, H] f32
    const float* w1 = (const float*)d_in[1];   // [E, F, H] f32
    const float* w2 = (const float*)d_in[2];   // [E, F, H] f32

    const int T = 8192, Hd = 1024, Fd = 4096, E = 8;
    const int cap = T / E;                     // 1024; BM=128 tiles never straddle experts

    __bf16* hws = (__bf16*)d_ws;               // [T, F] bf16 intermediate (64 MB)

    dim3 blk(256);
    moe_gemm1_gelu<<<dim3(Fd / BN, T / BM), blk, 0, stream>>>(x, w1, hws, Hd, Fd, cap);
    moe_gemm2<<<dim3(Hd / BN, T / BM), blk, 0, stream>>>(hws, w2, (float*)d_out,
                                                         Fd, Hd, cap, T);
}
